// DeformConv2d_53412213293881
// MI455X (gfx1250) — matrix operand order repaired and checked by reading, not compiled, not run on hardware
//
#include <hip/hip_runtime.h>
#include <cstdint>

// ---------------- problem constants ----------------
#define B_    4
#define CIN   64
#define COUT  64
#define H_    96
#define W_    96
#define PAD_  1
#define K2_   9
#define HW_   (H_ * W_)        // 9216
#define M_    (B_ * HW_)       // 36864  (divisible by 16; 9216 % 16 == 0 -> tiles never cross batch)
#define KDIM  (CIN * K2_)      // 576    (divisible by 32)

typedef _Float16 v8h  __attribute__((ext_vector_type(8)));
typedef _Float16 v16h __attribute__((ext_vector_type(16)));
typedef float    v8f  __attribute__((ext_vector_type(8)));
typedef float    v4f  __attribute__((ext_vector_type(4)));

// ---------------------------------------------------------------------------
// Kernel 1: pack weight (C_OUT, C_IN, 3, 3) f32 -> f16 Bmat[kc][o], kc = k*64+c
// ---------------------------------------------------------------------------
__global__ __launch_bounds__(256) void dcn_pack_w(const float* __restrict__ w,
                                                  _Float16* __restrict__ Bm) {
    int i  = blockIdx.x * 256 + threadIdx.x;   // grid covers exactly 576*64
    int kc = i >> 6;                            // 0..575
    int o  = i & 63;
    int k  = kc >> 6;                           // tap   0..8
    int c  = kc & 63;                           // chan  0..63
    Bm[i] = (_Float16)w[(size_t)o * (CIN * K2_) + c * K2_ + k];
}

// ---------------------------------------------------------------------------
// Kernel 2: mask = sigmoid(conv3x3(x, mask_weight) + bias)
// one thread per (b,h,w), mask_weight cached in LDS, 9 accumulators
// ---------------------------------------------------------------------------
__global__ __launch_bounds__(256) void dcn_mask(const float* __restrict__ x,
                                                const float* __restrict__ mw,
                                                const float* __restrict__ mb,
                                                float* __restrict__ mask) {
    __shared__ float smw[K2_ * CIN * K2_];      // 9*64*9 = 5184 floats = 20.7 KB
    for (int i = threadIdx.x; i < K2_ * CIN * K2_; i += 256) smw[i] = mw[i];
    __syncthreads();

    int idx = blockIdx.x * 256 + threadIdx.x;   // grid covers exactly B*HW = 36864
    int b  = idx / HW_;
    int hw = idx % HW_;
    int h  = hw / W_;
    int w  = hw % W_;
    const float* xb = x + (size_t)b * CIN * HW_;

    float acc[K2_];
#pragma unroll
    for (int k = 0; k < K2_; ++k) acc[k] = 0.f;

    for (int c = 0; c < CIN; ++c) {
        const float* xc = xb + c * HW_;
#pragma unroll
        for (int t = 0; t < K2_; ++t) {
            int yy = h - PAD_ + t / 3;
            int xx = w - PAD_ + t % 3;
            float xv = (yy >= 0 && yy < H_ && xx >= 0 && xx < W_) ? xc[yy * W_ + xx] : 0.f;
#pragma unroll
            for (int k = 0; k < K2_; ++k)
                acc[k] = fmaf(xv, smw[k * (CIN * K2_) + c * K2_ + t], acc[k]);
        }
    }
    float* mo = mask + (size_t)b * K2_ * HW_ + hw;
#pragma unroll
    for (int k = 0; k < K2_; ++k) {
        float z = acc[k] + mb[k];
        mo[k * HW_] = 1.f / (1.f + expf(-z));
    }
}

// ---------------------------------------------------------------------------
// Kernel 3: bilinear sample * mask -> f16 im2col A[m][k*64+c], m = b*HW + h*W + w
// one thread per (m, tap); validity & mask folded into corner weights
// ---------------------------------------------------------------------------
__global__ __launch_bounds__(256) void dcn_sample(const float* __restrict__ x,
                                                  const float* __restrict__ off,
                                                  const float* __restrict__ mask,
                                                  _Float16* __restrict__ Am) {
    int idx = blockIdx.x * 256 + threadIdx.x;   // grid covers exactly 9 * 36864
    int m = idx % M_;                            // adjacent threads -> adjacent pixels
    int k = idx / M_;
    int b  = m / HW_;
    int hw = m % HW_;
    int h  = hw / W_;
    int w  = hw % W_;
    int ky = k / 3, kx = k % 3;

    const float* offb = off + (size_t)b * (2 * K2_) * HW_;
    float oy = offb[(2 * k + 0) * HW_ + hw];
    float ox = offb[(2 * k + 1) * HW_ + hw];

    float py = (float)(h - PAD_ + ky) + oy;
    float px = (float)(w - PAD_ + kx) + ox;
    float y0f = floorf(py), x0f = floorf(px);
    int   y0  = (int)y0f,   x0  = (int)x0f;
    float wy = py - y0f, wx = px - x0f;

    float mval = mask[(size_t)b * K2_ * HW_ + (size_t)k * HW_ + hw];

    float w00 = (1.f - wy) * (1.f - wx) * mval;
    float w01 = (1.f - wy) * wx * mval;
    float w10 = wy * (1.f - wx) * mval;
    float w11 = wy * wx * mval;

    int y1 = y0 + 1, x1 = x0 + 1;
    bool vy0 = (y0 >= 0) & (y0 < H_), vy1 = (y1 >= 0) & (y1 < H_);
    bool vx0 = (x0 >= 0) & (x0 < W_), vx1 = (x1 >= 0) & (x1 < W_);
    if (!(vy0 & vx0)) w00 = 0.f;
    if (!(vy0 & vx1)) w01 = 0.f;
    if (!(vy1 & vx0)) w10 = 0.f;
    if (!(vy1 & vx1)) w11 = 0.f;

    int cy0 = min(max(y0, 0), H_ - 1), cy1 = min(max(y1, 0), H_ - 1);
    int cx0 = min(max(x0, 0), W_ - 1), cx1 = min(max(x1, 0), W_ - 1);
    int i00 = cy0 * W_ + cx0, i01 = cy0 * W_ + cx1;
    int i10 = cy1 * W_ + cx0, i11 = cy1 * W_ + cx1;

    const float* xb = x + (size_t)b * CIN * HW_;
    _Float16* arow = Am + (size_t)m * KDIM + k * CIN;
#pragma unroll 4
    for (int c = 0; c < CIN; ++c) {
        const float* xc = xb + c * HW_;
        float v = fmaf(xc[i00], w00,
                  fmaf(xc[i01], w01,
                  fmaf(xc[i10], w10, xc[i11] * w11)));
        arow[c] = (_Float16)v;
    }
}

// ---------------------------------------------------------------------------
// Kernel 4: C[M x 64] = A[M x 576] * B[576 x 64]  via v_wmma_f32_16x16x32_f16
// 256 threads = 8 wave32 waves; whole B (72 KB) staged in LDS once per block;
// each wave owns one 16-row M-tile and all 4 N-tiles (4 v8f accumulators).
// Output scattered into NCHW: out[b][o][hw], hw = m % 9216 (contiguous per lane).
// ---------------------------------------------------------------------------
#define SHUF16(lo, hi) __builtin_shufflevector(lo, hi, 0,1,2,3,4,5,6,7,8,9,10,11,12,13,14,15)

__global__ __launch_bounds__(256) void dcn_gemm(const _Float16* __restrict__ Am,
                                                const _Float16* __restrict__ Bm,
                                                float* __restrict__ out) {
    __shared__ _Float16 sB[KDIM * COUT];        // 576*64 f16 = 72 KB (WGP has 320 KB)

    const int tid = threadIdx.x;
    // cooperative 16B-chunk copy of the whole B matrix into LDS
    for (int i = tid; i < (KDIM * COUT) / 8; i += 256)
        ((v8h*)sB)[i] = ((const v8h*)Bm)[i];
    __syncthreads();

    const int wave = tid >> 5;
    const int lane = tid & 31;
    const int mtile = blockIdx.x * 8 + wave;    // grid = 288 blocks -> exactly 2304 tiles
    const int m0 = mtile * 16;

    // A-fragment addressing per ISA 16-bit A 16x32 layout:
    //   lanes 0-15  hold row M=lane,    K = {k0+0..7,  k0+16..23}
    //   lanes 16-31 hold row M=lane-16, K = {k0+8..15, k0+24..31}
    const int hs = lane >> 4;                   // 0 or 1
    const _Float16* arow = Am + (size_t)(m0 + (lane & 15)) * KDIM + hs * 8;

    v8f acc0 = {}, acc1 = {}, acc2 = {}, acc3 = {};

    for (int k0 = 0; k0 < KDIM; k0 += 32) {
        v8h alo = *(const v8h*)(arow + k0);
        v8h ahi = *(const v8h*)(arow + k0 + 16);
        v16h a = SHUF16(alo, ahi);

        // B-fragment: lane = N column (lane&15), K in same order as A: {8h+0..7, 16+8h+0..7}
        const _Float16* bcol = sB + (size_t)k0 * COUT + (lane & 15);
        v16h b0, b1, b2, b3;
#pragma unroll
        for (int s = 0; s < 16; ++s) {
            int kk = (s < 8) ? (hs * 8 + s) : (16 + hs * 8 + (s - 8));
            b0[s] = bcol[kk * COUT +  0];
            b1[s] = bcol[kk * COUT + 16];
            b2[s] = bcol[kk * COUT + 32];
            b3[s] = bcol[kk * COUT + 48];
        }

        acc0 = __builtin_amdgcn_wmma_f32_16x16x32_f16(false, a, false, b0, (short)0, acc0, false, false);
        acc1 = __builtin_amdgcn_wmma_f32_16x16x32_f16(false, a, false, b1, (short)0, acc1, false, false);
        acc2 = __builtin_amdgcn_wmma_f32_16x16x32_f16(false, a, false, b2, (short)0, acc2, false, false);
        acc3 = __builtin_amdgcn_wmma_f32_16x16x32_f16(false, a, false, b3, (short)0, acc3, false, false);
    }

    // C layout: VGPR r, lanes 0-15 -> (M=r, N=lane); lanes 16-31 -> (M=r+8, N=lane-16)
    // => this lane's 8 floats are 8 *consecutive* hw positions for one output channel.
    const int bb  = m0 / HW_;
    const int hw0 = (m0 % HW_) + hs * 8;
    const int n   = lane & 15;
    float* obase = out + (size_t)bb * COUT * HW_ + hw0;

    v8f accs[4] = {acc0, acc1, acc2, acc3};
#pragma unroll
    for (int nt = 0; nt < 4; ++nt) {
        float* op = obase + (size_t)(nt * 16 + n) * HW_;
        v4f lo = {accs[nt][0], accs[nt][1], accs[nt][2], accs[nt][3]};
        v4f hi = {accs[nt][4], accs[nt][5], accs[nt][6], accs[nt][7]};
        *(v4f*)(op + 0) = lo;
        *(v4f*)(op + 4) = hi;
    }
}

// ---------------------------------------------------------------------------
// launch
// ---------------------------------------------------------------------------
extern "C" void kernel_launch(void* const* d_in, const int* in_sizes, int n_in,
                              void* d_out, int out_size, void* d_ws, size_t ws_size,
                              hipStream_t stream) {
    const float* x   = (const float*)d_in[0];   // (4,64,96,96)
    const float* off = (const float*)d_in[1];   // (4,18,96,96)
    const float* wgt = (const float*)d_in[2];   // (64,64,3,3)
    const float* mw  = (const float*)d_in[3];   // (9,64,3,3)
    const float* mb  = (const float*)d_in[4];   // (9,)
    float* out = (float*)d_out;                 // (4,64,96,96)

    // workspace layout (all offsets 256B aligned):
    //   [0)                 mask  : B*K2*HW f32      = 1,327,104 B
    //   [1,327,104)         Bmat  : 576*64  f16      =    73,728 B
    //   [1,400,832)         Amat  : 36864*576 f16    = 42,467,328 B
    char* ws = (char*)d_ws;
    float*    maskbuf = (float*)ws;
    _Float16* Bmat    = (_Float16*)(ws + 1327104);
    _Float16* Amat    = (_Float16*)(ws + 1327104 + 73728);

    dcn_pack_w<<<(KDIM * COUT) / 256, 256, 0, stream>>>(wgt, Bmat);          // 144 blocks
    dcn_mask  <<<M_ / 256, 256, 0, stream>>>(x, mw, mb, maskbuf);            // 144 blocks
    dcn_sample<<<(K2_ * M_) / 256, 256, 0, stream>>>(x, off, maskbuf, Amat); // 1296 blocks
    dcn_gemm  <<<(M_ / 16) / 8, 256, 0, stream>>>(Amat, Bmat, out);          // 288 blocks
}
